// STLayer_66099546685511
// MI455X (gfx1250) — matrix operand order repaired and checked by reading, not compiled, not run on hardware
//
#include <hip/hip_runtime.h>
#include <hip/hip_bf16.h>
#include <stdint.h>

// ---------------------------------------------------------------------------
// Video-Swin block (shifted-window attention + MLP) for MI455X / gfx1250.
// All GEMMs routed through v_wmma_f32_16x16x32_f16 (f16 in, f32 accumulate).
// ---------------------------------------------------------------------------

typedef __attribute__((ext_vector_type(16))) _Float16 v16h;
typedef __attribute__((ext_vector_type(8)))  float    v8f;
typedef _Float16 half_t;

#define HWD   32      // spatial extent per axis
#define C_DIM 96
#define NH    6
#define HD    16
#define WS    8
#define NTOK  512     // tokens per window (8^3)
#define NWIN  64      // windows (4^3)
#define NTOTAL 32768  // 32^3 tokens
#define SHFT  4
#define QSCALE 0.25f  // 16^-0.5

__device__ __forceinline__ v8f vz8() {
  v8f a;
#pragma unroll
  for (int i = 0; i < 8; ++i) a[i] = 0.f;
  return a;
}

// A-fragment half index -> K (contiguous: lanes 0-15 K=0..15, lanes 16-31 K=16..31)
__device__ __forceinline__ int a_kmap(int h, int hi) {
  return 16 * hi + h;  // 16*hi + h
}

// A fragment from row-major f16 src (16 rows, 32 cols window, leading dim ld)
__device__ __forceinline__ v16h load_a_f16(const half_t* src, int ld) {
  const int lane = threadIdx.x & 31;
  const int m = lane & 15, hi = lane >> 4;
  v16h a;
#pragma unroll
  for (int h = 0; h < 16; ++h) a[h] = src[m * ld + a_kmap(h, hi)];
  return a;
}

// B fragment B[k][n] = w[n0+n][k0+k] from fp32 row-major weight [Nout][K].
// 16-bit B 32x16, contiguous k order chosen by this code: lanes 0-15 hold K=0..15 (n = lane), lanes 16-31 K=16..31.
__device__ __forceinline__ v16h load_b_wT(const float* w, int ld, int n0, int k0) {
  const int lane = threadIdx.x & 31;
  const int n = lane & 15, kb = (lane >> 4) << 4;
  const float* p = w + (size_t)(n0 + n) * ld + k0 + kb;
  v16h b;
#pragma unroll
  for (int h = 0; h < 16; ++h) b[h] = (half_t)p[h];
  return b;
}

__device__ __forceinline__ v8f wmma_f16(v16h a, v16h b, v8f c) {
  return __builtin_amdgcn_wmma_f32_16x16x32_f16(false, a, false, b,
                                                (short)0, c, false, false);
}

// shifted-window mask segment / group ids (computed analytically)
__device__ __forceinline__ int seg_id(int p) { return p < 24 ? 0 : (p < 28 ? 1 : 2); }
__device__ __forceinline__ int grp_id(int w, int i) {
  const int wh = w >> 4, ww = (w >> 2) & 3, wd = w & 3;
  const int ih = i >> 6, iw = (i >> 3) & 7, id = i & 7;
  return 9 * seg_id(wh * 8 + ih) + 3 * seg_id(ww * 8 + iw) + seg_id(wd * 8 + id);
}

// ---------------------------------------------------------------------------
// K1: LayerNorm1 + cyclic shift + window partition -> f16 WINA [NWIN][512][96]
// one wave per token (lane handles 3 channels), 8 tokens / 256-thread block
// ---------------------------------------------------------------------------
__global__ void k_ln1_window(const float* __restrict__ x,
                             const float* __restrict__ g,
                             const float* __restrict__ b,
                             half_t* __restrict__ wina) {
  const int token = blockIdx.x * 8 + (threadIdx.x >> 5);
  const int lane = threadIdx.x & 31;
  const int w = token >> 9, i = token & 511;
  const int wh = w >> 4, ww = (w >> 2) & 3, wd = w & 3;
  const int ih = i >> 6, iw = (i >> 3) & 7, id = i & 7;
  const int p0 = ((wh * 8 + ih) + SHFT) & 31;
  const int p1 = ((ww * 8 + iw) + SHFT) & 31;
  const int p2 = ((wd * 8 + id) + SHFT) & 31;
  const float* src = x + (((size_t)p0 * 32 + p1) * 32 + p2) * C_DIM;
  float v0 = src[lane], v1 = src[lane + 32], v2 = src[lane + 64];
  float s = v0 + v1 + v2, s2 = v0 * v0 + v1 * v1 + v2 * v2;
#pragma unroll
  for (int off = 16; off; off >>= 1) {
    s += __shfl_xor(s, off, 32);
    s2 += __shfl_xor(s2, off, 32);
  }
  const float mean = s * (1.f / 96.f);
  const float var = s2 * (1.f / 96.f) - mean * mean;
  const float inv = rsqrtf(var + 1e-5f);
  half_t* dst = wina + (size_t)token * C_DIM;
  dst[lane]      = (half_t)((v0 - mean) * inv * g[lane]      + b[lane]);
  dst[lane + 32] = (half_t)((v1 - mean) * inv * g[lane + 32] + b[lane + 32]);
  dst[lane + 64] = (half_t)((v2 - mean) * inv * g[lane + 64] + b[lane + 64]);
}

// ---------------------------------------------------------------------------
// K2: precompute relative-position-bias matrix biasM[6][512][512] (f32)
// ---------------------------------------------------------------------------
__global__ void k_bias(const float* __restrict__ rpb, float* __restrict__ biasM) {
  const int i = blockIdx.x, j = threadIdx.x;
  const int ih = i >> 6, iw = (i >> 3) & 7, id = i & 7;
  const int jh = j >> 6, jw = (j >> 3) & 7, jd = j & 7;
  const int idx = (ih - jh + 7) * 225 + (iw - jw + 7) * 15 + (id - jd + 7);
  const float* p = rpb + (size_t)idx * NH;
#pragma unroll
  for (int h = 0; h < NH; ++h)
    biasM[((size_t)h * 512 + i) * 512 + j] = p[h];
}

// ---------------------------------------------------------------------------
// K3: QKV GEMM (512x96 @ 96x288 per window) via WMMA; epilogue: bias, q-scale,
// scatter to per-head Q/K/V f16 buffers [w][head][tok][16].
// which/head derived from the wave-uniform tile index (no div/mod by 96).
// grid = 64*32 blocks, block = 128 (4 waves, each covers ~5 of 18 N-tiles)
// ---------------------------------------------------------------------------
__global__ void k_qkv(const half_t* __restrict__ wina,
                      const float* __restrict__ qkv_w,
                      const float* __restrict__ q_bias,
                      const float* __restrict__ v_bias,
                      half_t* __restrict__ Q, half_t* __restrict__ K,
                      half_t* __restrict__ V) {
  __shared__ half_t atile[16 * C_DIM];
  const int w = blockIdx.x >> 5, mt = blockIdx.x & 31;
  const int tid = threadIdx.x;
  __builtin_prefetch(qkv_w + (size_t)tid * 216, 0, 3);  // 288*96 floats spread
  const uint32_t* s32 = (const uint32_t*)(wina + ((size_t)w * NTOK + mt * 16) * C_DIM);
  uint32_t* d32 = (uint32_t*)atile;
#pragma unroll
  for (int t = tid; t < 768; t += 128) d32[t] = s32[t];
  __syncthreads();
  const int wv = tid >> 5, lane = tid & 31;
  const int n = lane & 15, hi = lane >> 4;
  for (int nt = wv; nt < 18; nt += 4) {
    v8f acc = vz8();
#pragma unroll
    for (int ks = 0; ks < 3; ++ks)
      acc = wmma_f16(load_a_f16(atile + ks * 32, C_DIM),
                     load_b_wT(qkv_w, C_DIM, nt * 16, ks * 32), acc);
    // nt 0-5 -> q head nt; 6-11 -> k head nt-6; 12-17 -> v head nt-12
    const int which = (nt >= 12) ? 2 : (nt >= 6 ? 1 : 0);   // wave-uniform
    const int head = nt - 6 * which;
    const int oc = head * 16 + n;
    const float bia = (which == 0) ? q_bias[oc] : (which == 2 ? v_bias[oc] : 0.f);
    const float mul = (which == 0) ? QSCALE : 1.f;
    half_t* outb = (which == 0) ? Q : (which == 1 ? K : V);
    half_t* dst = outb + (((size_t)w * NH + head) * NTOK + mt * 16) * HD + n;
#pragma unroll
    for (int r = 0; r < 8; ++r)
      dst[(size_t)(8 * hi + r) * HD] = (half_t)((acc[r] + bia) * mul);
  }
}

// ---------------------------------------------------------------------------
// K4: flash-style attention, one wave per (window, head, 16-query tile).
// QK^T via WMMA (K padded 16->32); bias + analytic shift mask; online softmax
// with width-16 butterfly shuffles; P transposed via LDS; PV with two key
// tiles per WMMA so K=32 is fully used.  grid = 64*6*32, block = 32
// (single-wave workgroup: barriers become S_NOP).
// ---------------------------------------------------------------------------
__global__ void k_attn(const half_t* __restrict__ Q, const half_t* __restrict__ K,
                       const half_t* __restrict__ V,
                       const float* __restrict__ biasM,
                       half_t* __restrict__ AO) {
  __shared__ half_t plds[16 * 32];
  const int bid = blockIdx.x;
  const int w = bid / (NH * 32), head = (bid / 32) % NH, mt = bid & 31;
  const int lane = threadIdx.x & 31;
  const int n = lane & 15, hi = lane >> 4;
  const size_t base = ((size_t)w * NH + head) * NTOK * HD;

  // Q A-fragment: K dims 0..15 valid (head dim), 16..31 zero
  v16h qf;
  {
    const half_t* qp = Q + base + (size_t)(mt * 16 + n) * HD;
#pragma unroll
    for (int h = 0; h < 16; ++h) {
      const int k = a_kmap(h, hi);
      qf[h] = (k < 16) ? qp[k] : (half_t)0;
    }
  }

  v8f oacc = vz8();
  float mrow[8], lrow[8];
  int grpQ[8];
#pragma unroll
  for (int r = 0; r < 8; ++r) {
    mrow[r] = -1e30f; lrow[r] = 0.f;
    grpQ[r] = grp_id(w, mt * 16 + 8 * hi + r);
  }
  const float* biasH = biasM + (size_t)head * 512 * 512;

  for (int jj = 0; jj < 16; ++jj) {
    v8f s[2];
#pragma unroll
    for (int t = 0; t < 2; ++t) {
      const int kt = jj * 2 + t;
      // B fragment from K: B[kdim=d][ncol=key]; only hi==0 lanes carry data
      v16h bk;
      const half_t* kp = K + base + (size_t)(kt * 16 + n) * HD;
#pragma unroll
      for (int h = 0; h < 16; ++h) bk[h] = (hi == 0) ? kp[h] : (half_t)0;
      v8f st = wmma_f16(qf, bk, vz8());
      const int jcol = kt * 16 + n;
      const int gk = grp_id(w, jcol);
      const float* bp = biasH + jcol;
#pragma unroll
      for (int r = 0; r < 8; ++r) {
        const int irow = mt * 16 + 8 * hi + r;
        st[r] += bp[(size_t)irow * 512] + ((grpQ[r] != gk) ? -100.f : 0.f);
      }
      s[t] = st;
    }
    // online softmax per row (rows split across half-waves; width-16 shuffles)
#pragma unroll
    for (int r = 0; r < 8; ++r) {
      float tm = fmaxf(s[0][r], s[1][r]);
#pragma unroll
      for (int off = 8; off; off >>= 1) tm = fmaxf(tm, __shfl_xor(tm, off, 16));
      const float nm = fmaxf(mrow[r], tm);
      const float corr = __expf(mrow[r] - nm);
      oacc[r] *= corr;
      const float p0 = __expf(s[0][r] - nm);
      const float p1 = __expf(s[1][r] - nm);
      float rs = p0 + p1;
#pragma unroll
      for (int off = 8; off; off >>= 1) rs += __shfl_xor(rs, off, 16);
      lrow[r] = lrow[r] * corr + rs;
      mrow[r] = nm;
      const int row = 8 * hi + r;
      plds[row * 32 + n]      = (half_t)p0;
      plds[row * 32 + 16 + n] = (half_t)p1;
    }
    __syncthreads();
    const v16h pf = load_a_f16(plds, 32);      // P as A-frag, K = 32 keys
    v16h bv;                                   // V as B-frag: B[key][d]
    const half_t* vp = V + base + (size_t)(jj * 32 + hi * 16) * HD + n;
#pragma unroll
    for (int h = 0; h < 16; ++h) bv[h] = vp[(size_t)h * HD];
    oacc = wmma_f16(pf, bv, oacc);
    __syncthreads();
  }
#pragma unroll
  for (int r = 0; r < 8; ++r) {
    const float o = oacc[r] / lrow[r];
    const int row = mt * 16 + 8 * hi + r;
    AO[((size_t)w * NTOK + row) * C_DIM + head * HD + n] = (half_t)o;
  }
}

// ---------------------------------------------------------------------------
// K5: proj GEMM + window reverse + inverse roll + residual -> X1 (f32)
// grid = 64*32, block = 192 (6 waves, one 16-col tile each)
// ---------------------------------------------------------------------------
__global__ void k_proj(const half_t* __restrict__ AO, const float* __restrict__ pw,
                       const float* __restrict__ pb, const float* __restrict__ x,
                       float* __restrict__ X1) {
  __shared__ half_t atile[16 * C_DIM];
  const int w = blockIdx.x >> 5, mt = blockIdx.x & 31;
  const int tid = threadIdx.x;
  __builtin_prefetch(pw + (size_t)tid * 48, 0, 3);      // 96*96 floats spread
  const uint32_t* s32 = (const uint32_t*)(AO + ((size_t)w * NTOK + mt * 16) * C_DIM);
  uint32_t* d32 = (uint32_t*)atile;
  for (int t = tid; t < 768; t += 192) d32[t] = s32[t];
  __syncthreads();
  const int nt = tid >> 5, lane = tid & 31;
  const int n = lane & 15, hi = lane >> 4;
  v8f acc = vz8();
#pragma unroll
  for (int ks = 0; ks < 3; ++ks)
    acc = wmma_f16(load_a_f16(atile + ks * 32, C_DIM),
                   load_b_wT(pw, C_DIM, nt * 16, ks * 32), acc);
  const int o = nt * 16 + n;
  const float bb = pb[o];
  const int wh = w >> 4, ww = (w >> 2) & 3, wd = w & 3;
#pragma unroll
  for (int r = 0; r < 8; ++r) {
    const int i = mt * 16 + 8 * hi + r;
    const int ih = i >> 6, iw = (i >> 3) & 7, id = i & 7;
    const int p0 = ((wh * 8 + ih) + SHFT) & 31;
    const int p1 = ((ww * 8 + iw) + SHFT) & 31;
    const int p2 = ((wd * 8 + id) + SHFT) & 31;
    const size_t idx = (((size_t)p0 * 32 + p1) * 32 + p2) * C_DIM + o;
    X1[idx] = x[idx] + acc[r] + bb;
  }
}

// ---------------------------------------------------------------------------
// K6: LayerNorm2 -> f16 H2 (natural layout)
// ---------------------------------------------------------------------------
__global__ void k_ln2(const float* __restrict__ X1, const float* __restrict__ g,
                      const float* __restrict__ b, half_t* __restrict__ H2) {
  const int token = blockIdx.x * 8 + (threadIdx.x >> 5);
  const int lane = threadIdx.x & 31;
  const float* src = X1 + (size_t)token * C_DIM;
  float v0 = src[lane], v1 = src[lane + 32], v2 = src[lane + 64];
  float s = v0 + v1 + v2, s2 = v0 * v0 + v1 * v1 + v2 * v2;
#pragma unroll
  for (int off = 16; off; off >>= 1) {
    s += __shfl_xor(s, off, 32);
    s2 += __shfl_xor(s2, off, 32);
  }
  const float mean = s * (1.f / 96.f);
  const float var = s2 * (1.f / 96.f) - mean * mean;
  const float inv = rsqrtf(var + 1e-5f);
  half_t* dst = H2 + (size_t)token * C_DIM;
  dst[lane]      = (half_t)((v0 - mean) * inv * g[lane]      + b[lane]);
  dst[lane + 32] = (half_t)((v1 - mean) * inv * g[lane + 32] + b[lane + 32]);
  dst[lane + 64] = (half_t)((v2 - mean) * inv * g[lane + 64] + b[lane + 64]);
}

// ---------------------------------------------------------------------------
// K7: FC1 (96 -> 384) + exact GELU -> f16 HID. grid = 2048, block = 256.
// ---------------------------------------------------------------------------
__global__ void k_fc1(const half_t* __restrict__ H2, const float* __restrict__ w1,
                      const float* __restrict__ b1, half_t* __restrict__ HID) {
  __shared__ half_t atile[16 * C_DIM];
  const int mt = blockIdx.x;
  const int tid = threadIdx.x;
  __builtin_prefetch(w1 + (size_t)tid * 144, 0, 3);     // 384*96 floats spread
  const uint32_t* s32 = (const uint32_t*)(H2 + (size_t)mt * 16 * C_DIM);
  uint32_t* d32 = (uint32_t*)atile;
#pragma unroll
  for (int t = tid; t < 768; t += 256) d32[t] = s32[t];
  __syncthreads();
  const int wv = tid >> 5, lane = tid & 31;
  const int n = lane & 15, hi = lane >> 4;
#pragma unroll
  for (int q = 0; q < 3; ++q) {
    const int nt = wv + q * 8;   // 24 N-tiles
    v8f acc = vz8();
#pragma unroll
    for (int ks = 0; ks < 3; ++ks)
      acc = wmma_f16(load_a_f16(atile + ks * 32, C_DIM),
                     load_b_wT(w1, C_DIM, nt * 16, ks * 32), acc);
    const int o = nt * 16 + n;
    const float bb = b1[o];
#pragma unroll
    for (int r = 0; r < 8; ++r) {
      const float h = acc[r] + bb;
      const float gel = 0.5f * h * (1.f + erff(h * 0.70710678118f));
      HID[((size_t)mt * 16 + 8 * hi + r) * 384 + o] = (half_t)gel;
    }
  }
}

// ---------------------------------------------------------------------------
// K8: FC2 (384 -> 96) + residual -> d_out (f32). grid = 2048, block = 192.
// ---------------------------------------------------------------------------
__global__ void k_fc2(const half_t* __restrict__ HID, const float* __restrict__ w2,
                      const float* __restrict__ b2, const float* __restrict__ X1,
                      float* __restrict__ out) {
  __shared__ half_t atile[16 * 384];
  const int mt = blockIdx.x;
  const int tid = threadIdx.x;
  __builtin_prefetch(w2 + (size_t)tid * 192, 0, 3);     // 96*384 floats spread
  const uint32_t* s32 = (const uint32_t*)(HID + (size_t)mt * 16 * 384);
  uint32_t* d32 = (uint32_t*)atile;
  for (int t = tid; t < 3072; t += 192) d32[t] = s32[t];
  __syncthreads();
  const int nt = tid >> 5, lane = tid & 31;
  const int n = lane & 15, hi = lane >> 4;
  v8f acc = vz8();
#pragma unroll
  for (int ks = 0; ks < 12; ++ks)
    acc = wmma_f16(load_a_f16(atile + ks * 32, 384),
                   load_b_wT(w2, 384, nt * 16, ks * 32), acc);
  const int o = nt * 16 + n;
  const float bb = b2[o];
#pragma unroll
  for (int r = 0; r < 8; ++r) {
    const size_t idx = ((size_t)mt * 16 + 8 * hi + r) * C_DIM + o;
    out[idx] = X1[idx] + acc[r] + bb;
  }
}

// ---------------------------------------------------------------------------
extern "C" void kernel_launch(void* const* d_in, const int* in_sizes, int n_in,
                              void* d_out, int out_size, void* d_ws, size_t ws_size,
                              hipStream_t stream) {
  const float* x     = (const float*)d_in[0];
  const float* n1g   = (const float*)d_in[1];
  const float* n1b   = (const float*)d_in[2];
  const float* qkvw  = (const float*)d_in[3];
  const float* qbia  = (const float*)d_in[4];
  const float* vbia  = (const float*)d_in[5];
  const float* rpb   = (const float*)d_in[6];
  const float* pw    = (const float*)d_in[7];
  const float* pb    = (const float*)d_in[8];
  const float* n2g   = (const float*)d_in[9];
  const float* n2b   = (const float*)d_in[10];
  const float* w1    = (const float*)d_in[11];
  const float* b1    = (const float*)d_in[12];
  const float* w2    = (const float*)d_in[13];
  const float* b2    = (const float*)d_in[14];

  char* ws = (char*)d_ws;
  size_t off = 0;
  half_t* WINA = (half_t*)(ws + off); off += (size_t)NTOTAL * C_DIM * 2;          // 6.3 MB
  half_t* Qb   = (half_t*)(ws + off); off += (size_t)NWIN * NH * NTOK * HD * 2;   // 6.3 MB
  half_t* Kb   = (half_t*)(ws + off); off += (size_t)NWIN * NH * NTOK * HD * 2;
  half_t* Vb   = (half_t*)(ws + off); off += (size_t)NWIN * NH * NTOK * HD * 2;
  float*  BIAS = (float*)(ws + off);  off += (size_t)NH * 512 * 512 * 4;          // 6.3 MB
  half_t* AO   = (half_t*)(ws + off); off += (size_t)NTOTAL * C_DIM * 2;
  float*  X1   = (float*)(ws + off);  off += (size_t)NTOTAL * C_DIM * 4;          // 12.6 MB
  half_t* H2   = (half_t*)(ws + off); off += (size_t)NTOTAL * C_DIM * 2;
  half_t* HID  = (half_t*)(ws + off); off += (size_t)NTOTAL * 384 * 2;            // 25.2 MB
  (void)off; (void)ws_size; (void)in_sizes; (void)n_in; (void)out_size;

  k_ln1_window<<<NTOTAL / 8, 256, 0, stream>>>(x, n1g, n1b, WINA);
  k_bias<<<512, 512, 0, stream>>>(rpb, BIAS);
  k_qkv<<<NWIN * 32, 128, 0, stream>>>(WINA, qkvw, qbia, vbia, Qb, Kb, Vb);
  k_attn<<<NWIN * NH * 32, 32, 0, stream>>>(Qb, Kb, Vb, BIAS, AO);
  k_proj<<<NWIN * 32, 192, 0, stream>>>(AO, pw, pb, x, X1);
  k_ln2<<<NTOTAL / 8, 256, 0, stream>>>(X1, n2g, n2b, H2);
  k_fc1<<<NTOTAL / 16, 256, 0, stream>>>(H2, w1, b1, HID);
  k_fc2<<<NTOTAL / 16, 192, 0, stream>>>(HID, w2, b2, X1, (float*)d_out);
}
